// MultiModalTransformerLayer_17927193493827
// MI455X (gfx1250) — compile-verified, matrix operand order checked by reading, not run on hardware
//
#include <hip/hip_runtime.h>
#include <hip/hip_bf16.h>
#include <stdint.h>
#include <stddef.h>

// ---------------------------------------------------------------------------
// Multimodal transformer layer for MI455X (gfx1250, wave32, WMMA bf16).
// ---------------------------------------------------------------------------

typedef __bf16 bf16_t;
typedef bf16_t bf16x16 __attribute__((ext_vector_type(16)));
typedef float  f32x8   __attribute__((ext_vector_type(8)));

union FragB { bf16x16 v; uint32_t u[8]; };

static constexpr int Bc  = 8;
static constexpr int Sc  = 1024;
static constexpr int Dc  = 1024;
static constexpr int Hc  = 8;
static constexpr int Mc  = 4;
static constexpr int HDc = 128;
static constexpr int FFNc = 4096;

__device__ __forceinline__ f32x8 wmma_bf16(const FragB& a, const FragB& b, f32x8 c) {
    return __builtin_amdgcn_wmma_f32_16x16x32_bf16(false, a.v, false, b.v,
                                                   (short)0, c, false, false);
}

__device__ __forceinline__ float gelu_exact(float x) {
    return 0.5f * x * (1.0f + erff(x * 0.70710678118654752f));
}

// ---------------------------------------------------------------------------
// fp32 -> bf16 elementwise (weight conversion)
// ---------------------------------------------------------------------------
__global__ __launch_bounds__(256) void k_f32_to_bf16(const float* __restrict__ src,
                                                     bf16_t* __restrict__ dst, int n) {
    int i = blockIdx.x * 256 + threadIdx.x;
    if (i < n) dst[i] = (bf16_t)src[i];
}

// ---------------------------------------------------------------------------
// LayerNorm over D=1024, one block per row, bf16 output
// ---------------------------------------------------------------------------
__global__ __launch_bounds__(256) void k_layernorm_bf16(const float* __restrict__ x,
                                                        const float* __restrict__ w,
                                                        const float* __restrict__ b,
                                                        bf16_t* __restrict__ out) {
    const int row = blockIdx.x, t = threadIdx.x;
    __shared__ float s1[256], s2[256];
    float v[4];
    float su = 0.f, sq = 0.f;
#pragma unroll
    for (int i = 0; i < 4; ++i) {
        float f = x[(size_t)row * Dc + t + i * 256];
        v[i] = f; su += f; sq += f * f;
    }
    s1[t] = su; s2[t] = sq;
    __syncthreads();
    for (int stp = 128; stp > 0; stp >>= 1) {
        if (t < stp) { s1[t] += s1[t + stp]; s2[t] += s2[t + stp]; }
        __syncthreads();
    }
    float mu   = s1[0] * (1.f / Dc);
    float var  = s2[0] * (1.f / Dc) - mu * mu;
    float rstd = rsqrtf(var + 1e-6f);
#pragma unroll
    for (int i = 0; i < 4; ++i) {
        int d = t + i * 256;
        out[(size_t)row * Dc + d] = (bf16_t)((v[i] - mu) * rstd * w[d] + b[d]);
    }
}

// ---------------------------------------------------------------------------
// Column mean over S: mean[b][d] = (1/S) sum_s xn[b][s][d]
// ---------------------------------------------------------------------------
__global__ __launch_bounds__(256) void k_colmean(const bf16_t* __restrict__ xn,
                                                 float* __restrict__ mean) {
    int d = blockIdx.x * 256 + threadIdx.x;
    int b = blockIdx.y;
    float s = 0.f;
    for (int si = 0; si < Sc; ++si)
        s += (float)xn[((size_t)(b * Sc + si)) * Dc + d];
    mean[b * Dc + d] = s * (1.f / Sc);
}

// ---------------------------------------------------------------------------
// Gate: softmax(mean @ Wg^T + bg) over 32 logits; keep 1 + 0.5*gate[b,h,M-1]
// One wave per batch (8 waves total).
// ---------------------------------------------------------------------------
__global__ __launch_bounds__(256) void k_gate(const float* __restrict__ mean,
                                              const float* __restrict__ Wg,
                                              const float* __restrict__ bg,
                                              float* __restrict__ row_scale) {
    int t = threadIdx.x;
    int b = t >> 5, j = t & 31;
    float acc = bg[j];
    for (int d = 0; d < Dc; ++d) acc += mean[b * Dc + d] * Wg[j * Dc + d];
    float mx = acc;
#pragma unroll
    for (int off = 16; off > 0; off >>= 1) mx = fmaxf(mx, __shfl_xor(mx, off, 32));
    float e = __expf(acc - mx);
    float sum = e;
#pragma unroll
    for (int off = 16; off > 0; off >>= 1) sum += __shfl_xor(sum, off, 32);
    float g = e / sum;
    if ((j & 3) == 3) row_scale[b * Hc + (j >> 2)] = 1.f + 0.5f * g;
}

// ---------------------------------------------------------------------------
// Precompute mask vectors: maskk[b][s]=mm, maskq[b][s]=mm*(s<H ? rs[b][s] : 1)
// ---------------------------------------------------------------------------
__global__ __launch_bounds__(256) void k_mask(const int* __restrict__ mids,
                                              const float* __restrict__ row_scale,
                                              float* __restrict__ maskq,
                                              float* __restrict__ maskk) {
    int idx = blockIdx.x * 256 + threadIdx.x;
    int b = idx / Sc, s = idx % Sc;
    float mm = (mids[idx] == (Mc - 1)) ? 1.f : 0.f;
    maskk[idx] = mm;
    float rs = (s < Hc) ? row_scale[b * Hc + s] : 1.f;
    maskq[idx] = mm * rs;
}

// ---------------------------------------------------------------------------
// Tiled WMMA GEMM: out[row][n] = A[row][:] . Wt[n][:] + bias[n]  (+ epilogue)
//   MODE 0: bf16 out        MODE 1: gelu -> bf16 out     MODE 2: f32 out + resid
// Block tile 128(M) x 128(N), BK=32, 8 waves each doing a 32x64 micro-tile
// (2 A-frags x 4 B-frags = 8 WMMAs per k-step -> 6 LDS dwords per WMMA).
// LDS rows padded to 34 bf16 (17 dwords) to avoid bank conflicts.
// ---------------------------------------------------------------------------
template <int MODE>
__global__ __launch_bounds__(256) void k_gemm_wmma(const bf16_t* __restrict__ A,
                                                   const bf16_t* __restrict__ Wt,
                                                   const float* __restrict__ bias,
                                                   const float* __restrict__ resid,
                                                   bf16_t* __restrict__ outb,
                                                   float* __restrict__ outf,
                                                   int K, int N) {
    const int t = threadIdx.x;
    const int w = t >> 5, lane = t & 31;
    const int l15 = lane & 15, lh = lane >> 4;
    const int kb2 = lh * 4;
    const int wm = w & 3, wn = w >> 2;            // 4 row-groups x 2 col-groups
    const int rowBase = blockIdx.y * 128;
    const int colBase = blockIdx.x * 128;

    __shared__ uint32_t AsU[128 * 17];
    __shared__ uint32_t BsU[128 * 17];

    f32x8 acc[2][4];
#pragma unroll
    for (int i = 0; i < 2; ++i)
#pragma unroll
        for (int j = 0; j < 4; ++j)
#pragma unroll
            for (int r = 0; r < 8; ++r) acc[i][j][r] = 0.f;

    const uint32_t* Agp = (const uint32_t*)A;
    const uint32_t* Wgp = (const uint32_t*)Wt;

    for (int kk = 0; kk < K; kk += 32) {
#pragma unroll
        for (int i = 0; i < 8; ++i) {               // A tile: 128 rows x 16 dwords
            int lin = i * 256 + t;
            int row = lin >> 4, col = lin & 15;
            AsU[row * 17 + col] = Agp[((size_t)(rowBase + row) * K + kk) / 2 + col];
        }
#pragma unroll
        for (int i = 0; i < 8; ++i) {               // W tile: 128 rows x 16 dwords
            int lin = i * 256 + t;
            int row = lin >> 4, col = lin & 15;
            BsU[row * 17 + col] = Wgp[((size_t)(colBase + row) * K + kk) / 2 + col];
        }
        if (kk + 32 < K)
            __builtin_prefetch(A + (size_t)(rowBase + (t >> 1)) * K + kk + 32, 0, 1);
        __syncthreads();

        FragB af[2], bfv[4];
#pragma unroll
        for (int i = 0; i < 2; ++i) {
            int rb = (wm * 32 + i * 16 + l15) * 17;
#pragma unroll
            for (int p = 0; p < 4; ++p) {
                af[i].u[p]     = AsU[rb + kb2 + p];
                af[i].u[4 + p] = AsU[rb + kb2 + 8 + p];
            }
        }
#pragma unroll
        for (int j = 0; j < 4; ++j) {
            int rb = (wn * 64 + j * 16 + l15) * 17;
#pragma unroll
            for (int p = 0; p < 4; ++p) {
                bfv[j].u[p]     = BsU[rb + kb2 + p];
                bfv[j].u[4 + p] = BsU[rb + kb2 + 8 + p];
            }
        }
#pragma unroll
        for (int i = 0; i < 2; ++i)
#pragma unroll
            for (int j = 0; j < 4; ++j)
                acc[i][j] = wmma_bf16(af[i], bfv[j], acc[i][j]);
        __syncthreads();
    }

#pragma unroll
    for (int i = 0; i < 2; ++i) {
#pragma unroll
        for (int j = 0; j < 4; ++j) {
            int colG = colBase + wn * 64 + j * 16 + l15;
            float bj = bias[colG];
#pragma unroll
            for (int r = 0; r < 8; ++r) {
                int rowG = rowBase + wm * 32 + i * 16 + r + lh * 8;
                float vv = acc[i][j][r] + bj;
                if (MODE == 1) vv = gelu_exact(vv);
                if (MODE == 2) {
                    outf[(size_t)rowG * N + colG] = vv + resid[(size_t)rowG * N + colG];
                } else {
                    outb[(size_t)rowG * N + colG] = (bf16_t)vv;
                }
            }
        }
    }
}

// ---------------------------------------------------------------------------
// Single-pass flash attention with modality mask.
// grid = (S/128, H, B), block = 256 (8 waves). Each wave: 16 q-rows.
// K staged in LDS row-major; V staged transposed; P bounced via per-wave LDS.
// ---------------------------------------------------------------------------
__global__ __launch_bounds__(256) void k_attention(const bf16_t* __restrict__ qb,
                                                   const bf16_t* __restrict__ kbuf,
                                                   const bf16_t* __restrict__ vbuf,
                                                   const float* __restrict__ maskq,
                                                   const float* __restrict__ maskk,
                                                   bf16_t* __restrict__ ob) {
    const int sQ = blockIdx.x * 128;
    const int h  = blockIdx.y;
    const int b  = blockIdx.z;
    const int t = threadIdx.x, w = t >> 5, lane = t & 31;
    const int l15 = lane & 15, lh = lane >> 4;
    const int kb2 = lh * 4;
    const int qRow0 = sQ + w * 16;
    const float scale = 0.08838834764831845f;  // 1/sqrt(128)

    __shared__ uint32_t KsU[32 * 65];                    // K chunk: 32 x 128 bf16
    __shared__ __align__(16) bf16_t VtB[128 * 34];       // V^T chunk: 128 x 32 bf16
    __shared__ __align__(16) bf16_t PsB[8 * 16 * 34];    // per-wave P staging
    uint32_t* VtU = (uint32_t*)VtB;
    uint32_t* PsU = (uint32_t*)PsB;

    // Load this wave's Q fragments (16 rows x 128 d = 4 A-frags) from global.
    FragB qf[4];
    {
        const uint32_t* qg = (const uint32_t*)qb;
        size_t base = ((size_t)(b * Sc + qRow0 + l15) * Dc + h * HDc) / 2;
#pragma unroll
        for (int f = 0; f < 4; ++f) {
            size_t fb = base + f * 16;
#pragma unroll
            for (int p = 0; p < 4; ++p) {
                qf[f].u[p]     = qg[fb + kb2 + p];
                qf[f].u[4 + p] = qg[fb + kb2 + 8 + p];
            }
        }
    }
    float mq[8];
#pragma unroll
    for (int r = 0; r < 8; ++r) mq[r] = maskq[b * Sc + qRow0 + r + lh * 8];

    f32x8 acc[8];
#pragma unroll
    for (int dt = 0; dt < 8; ++dt)
#pragma unroll
        for (int r = 0; r < 8; ++r) acc[dt][r] = 0.f;
    float mrow[8], lrow[8];
#pragma unroll
    for (int r = 0; r < 8; ++r) { mrow[r] = -1e30f; lrow[r] = 0.f; }

    for (int c = 0; c < Sc / 32; ++c) {
        const int s0 = c * 32;
        {   // cooperative stage of K (row-major) and V (transposed)
            const uint32_t* kg = (const uint32_t*)kbuf;
            const uint32_t* vg = (const uint32_t*)vbuf;
#pragma unroll
            for (int i = 0; i < 8; ++i) {
                int lin = i * 256 + t;
                int row = lin >> 6, col = lin & 63;
                KsU[row * 65 + col] =
                    kg[((size_t)(b * Sc + s0 + row) * Dc + h * HDc) / 2 + col];
            }
#pragma unroll
            for (int i = 0; i < 8; ++i) {
                int lin = i * 256 + t;
                int row = lin >> 6, col = lin & 63;
                uint32_t vv =
                    vg[((size_t)(b * Sc + s0 + row) * Dc + h * HDc) / 2 + col];
                int d0 = col * 2;
                VtB[d0 * 34 + row] =
                    __builtin_bit_cast(bf16_t, (unsigned short)(vv & 0xFFFFu));
                VtB[(d0 + 1) * 34 + row] =
                    __builtin_bit_cast(bf16_t, (unsigned short)(vv >> 16));
            }
        }
        __syncthreads();

        // Scores: two 16x16 C-tiles, K-dim = 128 (4 WMMAs each)
        f32x8 sc[2];
#pragma unroll
        for (int st = 0; st < 2; ++st) {
            f32x8 cz;
#pragma unroll
            for (int r = 0; r < 8; ++r) cz[r] = 0.f;
            int srow = st * 16 + l15;
#pragma unroll
            for (int kd = 0; kd < 4; ++kd) {
                FragB kf;
                int rb = srow * 65 + kd * 16;
#pragma unroll
                for (int p = 0; p < 4; ++p) {
                    kf.u[p]     = KsU[rb + kb2 + p];
                    kf.u[4 + p] = KsU[rb + kb2 + 8 + p];
                }
                cz = wmma_bf16(qf[kd], kf, cz);
            }
            sc[st] = cz;
        }

        float mk0 = maskk[b * Sc + s0 + l15];
        float mk1 = maskk[b * Sc + s0 + 16 + l15];
        float cmax[8];
#pragma unroll
        for (int r = 0; r < 8; ++r) {
            float a0 = sc[0][r] * scale + mq[r] * mk0;
            float a1 = sc[1][r] * scale + mq[r] * mk1;
            sc[0][r] = a0; sc[1][r] = a1;
            cmax[r] = fmaxf(a0, a1);
        }
#pragma unroll
        for (int off = 1; off < 16; off <<= 1)
#pragma unroll
            for (int r = 0; r < 8; ++r)
                cmax[r] = fmaxf(cmax[r], __shfl_xor(cmax[r], off, 32));

        float alpha[8];
#pragma unroll
        for (int r = 0; r < 8; ++r) {
            float mn = fmaxf(mrow[r], cmax[r]);
            alpha[r] = __expf(mrow[r] - mn);
            mrow[r] = mn;
        }
        // exp(P) -> per-wave LDS staging (C layout -> A layout bounce)
#pragma unroll
        for (int r = 0; r < 8; ++r) {
            float p0 = __expf(sc[0][r] - mrow[r]);
            float p1 = __expf(sc[1][r] - mrow[r]);
            lrow[r] = lrow[r] * alpha[r] + p0 + p1;
            int m = r + lh * 8;
            PsB[w * 16 * 34 + m * 34 + l15]      = (bf16_t)p0;
            PsB[w * 16 * 34 + m * 34 + 16 + l15] = (bf16_t)p1;
        }
        FragB pf;
        {
            int rb = w * 16 * 17 + l15 * 17;
#pragma unroll
            for (int p = 0; p < 4; ++p) {
                pf.u[p]     = PsU[rb + kb2 + p];
                pf.u[4 + p] = PsU[rb + kb2 + 8 + p];
            }
        }
        // O += P @ V  (rescale accumulators by alpha first)
#pragma unroll
        for (int dt = 0; dt < 8; ++dt) {
            f32x8 a = acc[dt];
#pragma unroll
            for (int r = 0; r < 8; ++r) a[r] *= alpha[r];
            FragB vf;
            int rb = (dt * 16 + l15) * 17;
#pragma unroll
            for (int p = 0; p < 4; ++p) {
                vf.u[p]     = VtU[rb + kb2 + p];
                vf.u[4 + p] = VtU[rb + kb2 + 8 + p];
            }
            acc[dt] = wmma_bf16(pf, vf, a);
        }
        __syncthreads();
    }

    // finalize: reduce l across the 16-lane half, normalize, store bf16
#pragma unroll
    for (int off = 1; off < 16; off <<= 1)
#pragma unroll
        for (int r = 0; r < 8; ++r) lrow[r] += __shfl_xor(lrow[r], off, 32);
#pragma unroll
    for (int dt = 0; dt < 8; ++dt) {
#pragma unroll
        for (int r = 0; r < 8; ++r) {
            int q = qRow0 + r + lh * 8;
            int d = dt * 16 + l15;
            float o = acc[dt][r] / lrow[r];
            ob[(size_t)(b * Sc + q) * Dc + h * HDc + d] = (bf16_t)o;
        }
    }
}

// ---------------------------------------------------------------------------
// Host-side orchestration
// ---------------------------------------------------------------------------
extern "C" void kernel_launch(void* const* d_in, const int* in_sizes, int n_in,
                              void* d_out, int out_size, void* d_ws, size_t ws_size,
                              hipStream_t stream) {
    (void)in_sizes; (void)n_in; (void)out_size; (void)ws_size;

    const float* x    = (const float*)d_in[0];
    const int*   mids = (const int*)d_in[1];
    const float* ln1w = (const float*)d_in[2];
    const float* ln1b = (const float*)d_in[3];
    const float* Wq   = (const float*)d_in[4];
    const float* bq   = (const float*)d_in[5];
    const float* Wk   = (const float*)d_in[6];
    const float* bk   = (const float*)d_in[7];
    const float* Wv   = (const float*)d_in[8];
    const float* bv   = (const float*)d_in[9];
    const float* Wo   = (const float*)d_in[10];
    const float* bo   = (const float*)d_in[11];
    const float* Wg   = (const float*)d_in[12];
    const float* bg   = (const float*)d_in[13];
    const float* ln2w = (const float*)d_in[14];
    const float* ln2b = (const float*)d_in[15];
    const float* W1   = (const float*)d_in[16];
    const float* b1   = (const float*)d_in[17];
    const float* W2   = (const float*)d_in[18];
    const float* b2   = (const float*)d_in[19];
    float* out = (float*)d_out;

    const size_t Rows = (size_t)Bc * Sc;      // 8192
    char* ws = (char*)d_ws;
    size_t off = 0;
    auto carve = [&](size_t bytes) -> void* {
        void* p = ws + off;
        off += (bytes + 255) & ~(size_t)255;
        return p;
    };
    bf16_t* wq_b  = (bf16_t*)carve((size_t)Dc * Dc * 2);
    bf16_t* wk_b  = (bf16_t*)carve((size_t)Dc * Dc * 2);
    bf16_t* wv_b  = (bf16_t*)carve((size_t)Dc * Dc * 2);
    bf16_t* wo_b  = (bf16_t*)carve((size_t)Dc * Dc * 2);
    bf16_t* w1_b  = (bf16_t*)carve((size_t)FFNc * Dc * 2);
    bf16_t* w2_b  = (bf16_t*)carve((size_t)Dc * FFNc * 2);
    bf16_t* xn_b  = (bf16_t*)carve(Rows * Dc * 2);       // reused for xn2
    bf16_t* q_b   = (bf16_t*)carve(Rows * Dc * 2);
    bf16_t* k_b   = (bf16_t*)carve(Rows * Dc * 2);
    bf16_t* v_b   = (bf16_t*)carve(Rows * Dc * 2);
    bf16_t* ao_b  = (bf16_t*)carve(Rows * Dc * 2);
    float*  x1_f  = (float*)carve(Rows * Dc * 4);
    bf16_t* h_b   = (bf16_t*)carve(Rows * FFNc * 2);
    float*  meanp = (float*)carve((size_t)Bc * Dc * 4);
    float*  rsp   = (float*)carve((size_t)Bc * Hc * 4);
    float*  mqp   = (float*)carve(Rows * 4);
    float*  mkp   = (float*)carve(Rows * 4);

    // --- weight conversion to bf16 ---
    int nDD = Dc * Dc, nFD = FFNc * Dc;
    k_f32_to_bf16<<<(nDD + 255) / 256, 256, 0, stream>>>(Wq, wq_b, nDD);
    k_f32_to_bf16<<<(nDD + 255) / 256, 256, 0, stream>>>(Wk, wk_b, nDD);
    k_f32_to_bf16<<<(nDD + 255) / 256, 256, 0, stream>>>(Wv, wv_b, nDD);
    k_f32_to_bf16<<<(nDD + 255) / 256, 256, 0, stream>>>(Wo, wo_b, nDD);
    k_f32_to_bf16<<<(nFD + 255) / 256, 256, 0, stream>>>(W1, w1_b, nFD);
    k_f32_to_bf16<<<(nFD + 255) / 256, 256, 0, stream>>>(W2, w2_b, nFD);

    // --- LN1, gate, masks ---
    k_layernorm_bf16<<<(int)Rows, 256, 0, stream>>>(x, ln1w, ln1b, xn_b);
    k_colmean<<<dim3(Dc / 256, Bc), 256, 0, stream>>>(xn_b, meanp);
    k_gate<<<1, 256, 0, stream>>>(meanp, Wg, bg, rsp);
    k_mask<<<(int)(Rows / 256), 256, 0, stream>>>(mids, rsp, mqp, mkp);

    // --- QKV projections (bf16 out) ---
    dim3 gD(Dc / 128, (unsigned)(Rows / 128));
    k_gemm_wmma<0><<<gD, 256, 0, stream>>>(xn_b, wq_b, bq, nullptr, q_b, nullptr, Dc, Dc);
    k_gemm_wmma<0><<<gD, 256, 0, stream>>>(xn_b, wk_b, bk, nullptr, k_b, nullptr, Dc, Dc);
    k_gemm_wmma<0><<<gD, 256, 0, stream>>>(xn_b, wv_b, bv, nullptr, v_b, nullptr, Dc, Dc);

    // --- flash attention ---
    k_attention<<<dim3(Sc / 128, Hc, Bc), 256, 0, stream>>>(q_b, k_b, v_b, mqp, mkp, ao_b);

    // --- output projection + residual (f32) ---
    k_gemm_wmma<2><<<gD, 256, 0, stream>>>(ao_b, wo_b, bo, x, nullptr, x1_f, Dc, Dc);

    // --- LN2, FFN ---
    k_layernorm_bf16<<<(int)Rows, 256, 0, stream>>>(x1_f, ln2w, ln2b, xn_b);
    dim3 gF(FFNc / 128, (unsigned)(Rows / 128));
    k_gemm_wmma<1><<<gF, 256, 0, stream>>>(xn_b, w1_b, b1, nullptr, h_b, nullptr, Dc, FFNc);
    k_gemm_wmma<2><<<gD, 256, 0, stream>>>(h_b, w2_b, b2, x1_f, nullptr, out, FFNc, Dc);
}
